// SelfAttention_vParam_6485400617221
// MI455X (gfx1250) — matrix operand order repaired and checked by reading, not compiled, not run on hardware
//
#include <hip/hip_runtime.h>

// ---------------------------------------------------------------------------
// Self-attention on MI455X (gfx1250, wave32, WMMA + async global->LDS DMA).
//   x:[4,2048,1024] f32, W_q/W_k/W_v:[1024,1024] f32 -> out:[4,2048,1024] f32
//
// Phases:
//   0a) x  -> xh  bf16                       (elementwise convert)
//   0b) W  -> Wt  bf16, transposed [dout][din]  (so GEMM B is always [N][K])
//   1)  Q = xh@W, K = xh@W  (bf16 out);  V written TRANSPOSED: Vt[d][key]
//   2)  S = (Q@K^T) * 1/32  (bf16 scores)     -- B = K[key][d] is [N][K]
//   3)  row softmax in-place (f32 math, wave32 shuffles)
//   4)  out = P@V (f32)                       -- B = Vt[d][key] is [N][K]
//
// GEMM: 256 threads = 8 waves, block tile 256(M) x 64(N), BK=32.
// Each wave: 32 rows x 64 cols = 8 accumulators -> 8 v_wmma per k-step.
// A/B tiles staged with GLOBAL_LOAD_ASYNC_TO_LDS_B128 (ASYNCcnt), double
// buffered: prefetch tile k+1 overlaps WMMA on tile k; one barrier/step.
// All 12 fragment ds_load_b128 are issued before the 8 WMMAs so the matrix
// pipe drains back-to-back under partial s_wait_dscnt.
// Working set (~102 MB ws) fits the 192 MB L2; compute-bound (~120 GFLOP).
// ---------------------------------------------------------------------------

typedef __attribute__((ext_vector_type(16))) __bf16        v16bf;
typedef __attribute__((ext_vector_type(8)))  float          v8f;
typedef __attribute__((ext_vector_type(8)))  unsigned short v8u;

union FragU {            // two 16B LDS reads -> one WMMA bf16 operand
  v8u   u[2];
  v16bf b;
};

__device__ __forceinline__ unsigned short f32_to_bf16_rne(float f) {
  unsigned int u = __float_as_uint(f);
  u += 0x7FFFu + ((u >> 16) & 1u);
  return (unsigned short)(u >> 16);
}
__device__ __forceinline__ float bf16_to_f32(unsigned short h) {
  return __uint_as_float(((unsigned int)h) << 16);
}

#define BM 256
#define BN 64
#define BK 32
#define LDS_PITCH 40   // 80-byte rows: 16B aligned, 20-bank stride

// Issue one 16-byte async DMA: global -> LDS (GV mode, ASYNCcnt-tracked).
__device__ __forceinline__ void async_b128(void* lds_ptr, const void* gptr) {
  const unsigned ldsa = (unsigned)(unsigned long long)lds_ptr;  // low 32 = LDS offset
  const unsigned long long ga = (unsigned long long)gptr;
  asm volatile("global_load_async_to_lds_b128 %0, %1, off"
               :: "v"(ldsa), "v"(ga) : "memory");
}
__device__ __forceinline__ void wait_async0() {
  asm volatile("s_wait_asynccnt 0" ::: "memory");
}

// OUTMODE: 0 = f32, 1 = bf16, 2 = bf16 transposed (C[col][row])
template <int OUTMODE>
__global__ __launch_bounds__(256, 2) void gemm_bf16_wmma(
    const unsigned short* __restrict__ A,   // [M][K] bf16, row-major
    const unsigned short* __restrict__ B,   // [N][K] bf16, row-major (N-major)
    void* __restrict__ C,
    int Ktot, int lda, int ldb, int ldc,
    long sA, long sB, long sC, float scale)
{
  __shared__ unsigned short lsA[2][BM][LDS_PITCH];
  __shared__ unsigned short lsB[2][BN][LDS_PITCH];

  const int t    = threadIdx.x;
  const int lane = t & 31;
  const int wv   = t >> 5;        // wave 0..7 -> 32-row strip
  const int lrow = lane & 15;
  const int hf   = lane >> 4;     // half-wave select (ISA fragment layout)

  const int m0 = blockIdx.y * BM;
  const int n0 = blockIdx.x * BN;

  const unsigned short* Ah = A + (size_t)blockIdx.z * sA;
  const unsigned short* Bh = B + (size_t)blockIdx.z * sB;

  v8f acc0[4] = {};
  v8f acc1[4] = {};

  // ---- async stage of one (A,B) k-tile into buffer `buf` -------------------
  auto prefetch = [&](int buf, int k0) {
    // A: BM x BK = 1024 x 16B chunks -> 4 per thread
#pragma unroll
    for (int it = 0; it < (BM * BK) / (256 * 8); ++it) {
      const int c   = t + it * 256;
      const int row = c >> 2;            // BK/8 == 4 chunks per row
      const int kk  = (c & 3) * 8;
      async_b128(&lsA[buf][row][kk],
                 Ah + (size_t)(m0 + row) * lda + k0 + kk);
    }
    // B: BN x BK = 256 x 16B chunks -> 1 per thread
    {
      const int row = t >> 2;
      const int kk  = (t & 3) * 8;
      async_b128(&lsB[buf][row][kk],
                 Bh + (size_t)(n0 + row) * ldb + k0 + kk);
    }
  };

  // ---- WMMA on buffer `buf`: issue ALL ds loads, then all 8 WMMAs ---------
  auto compute = [&](int buf) {
    // A fragment (16x32 bf16): lane L -> row L%16;
    // elems 0..7: K=(L/16)*8+i ; elems 8..15: K=16+(L/16)*8+i
    const unsigned short* ar0 = &lsA[buf][wv * 32 + lrow][0];
    const unsigned short* ar1 = &lsA[buf][wv * 32 + 16 + lrow][0];
    const unsigned short* br0 = &lsB[buf][lrow][0];
    FragU fa0, fa1, fb[4];
    fa0.u[0] = *(const v8u*)(ar0 + hf * 8);
    fa0.u[1] = *(const v8u*)(ar0 + 16 + hf * 8);
    fa1.u[0] = *(const v8u*)(ar1 + hf * 8);
    fa1.u[1] = *(const v8u*)(ar1 + 16 + hf * 8);
#pragma unroll
    for (int ns = 0; ns < 4; ++ns) {
      // B fragment (32x16): lane L -> col L%16; elems 0..7: K=(L/16)*8+i ; elems 8..15: K=16+(L/16)*8+i
      const unsigned short* br = br0 + ns * 16 * LDS_PITCH;
      fb[ns].u[0] = *(const v8u*)(br + hf * 8);
      fb[ns].u[1] = *(const v8u*)(br + 16 + hf * 8);
    }
#pragma unroll
    for (int ns = 0; ns < 4; ++ns)
      acc0[ns] = __builtin_amdgcn_wmma_f32_16x16x32_bf16(
          false, fa0.b, false, fb[ns].b, (short)0, acc0[ns], false, false);
#pragma unroll
    for (int ns = 0; ns < 4; ++ns)
      acc1[ns] = __builtin_amdgcn_wmma_f32_16x16x32_bf16(
          false, fa1.b, false, fb[ns].b, (short)0, acc1[ns], false, false);
  };

  // ---- double-buffered main loop ------------------------------------------
  prefetch(0, 0);
  int k0 = 0;
  while (true) {
    wait_async0();
    __syncthreads();
    if (k0 + BK < Ktot) prefetch(1, k0 + BK);
    compute(0);
    k0 += BK;
    if (k0 >= Ktot) break;

    wait_async0();
    __syncthreads();
    if (k0 + BK < Ktot) prefetch(0, k0 + BK);
    compute(1);
    k0 += BK;
    if (k0 >= Ktot) break;
  }

  // ---- epilogue: D layout VGPR r -> row (L/16)*8 + r, col L%16 -------------
  const size_t zC = (size_t)blockIdx.z * sC;
#pragma unroll
  for (int ms = 0; ms < 2; ++ms) {
    const v8f* acc = ms ? acc1 : acc0;
#pragma unroll
    for (int ns = 0; ns < 4; ++ns) {
      const int col = n0 + ns * 16 + lrow;
#pragma unroll
      for (int r = 0; r < 8; ++r) {
        const int row = m0 + wv * 32 + ms * 16 + hf * 8 + r;
        const float v = acc[ns][r] * scale;
        if (OUTMODE == 0) {
          ((float*)C)[zC + (size_t)row * ldc + col] = v;
        } else if (OUTMODE == 1) {
          ((unsigned short*)C)[zC + (size_t)row * ldc + col] = f32_to_bf16_rne(v);
        } else {
          ((unsigned short*)C)[zC + (size_t)col * ldc + row] = f32_to_bf16_rne(v);
        }
      }
    }
  }
}

// ---------------------------------------------------------------------------
// Elementwise f32 -> bf16 (vectorized 4/thread).
__global__ __launch_bounds__(256) void conv_f32_bf16(
    const float* __restrict__ in, unsigned short* __restrict__ out)
{
  const size_t i = ((size_t)blockIdx.x * 256 + threadIdx.x) * 4;
  const float4 v = *(const float4*)(in + i);
  uint2 o;
  o.x = (unsigned)f32_to_bf16_rne(v.x) | ((unsigned)f32_to_bf16_rne(v.y) << 16);
  o.y = (unsigned)f32_to_bf16_rne(v.z) | ((unsigned)f32_to_bf16_rne(v.w) << 16);
  *(uint2*)(out + i) = o;
}

// f32 [K][N] -> bf16 transposed [N][K] via 32x32 LDS tile.
__global__ __launch_bounds__(256) void transpose_conv_f32_bf16(
    const float* __restrict__ W, unsigned short* __restrict__ Wt, int K, int N)
{
  __shared__ float tile[32][33];
  const int k0 = blockIdx.y * 32;
  const int n0 = blockIdx.x * 32;
  const int tx = threadIdx.x & 31;
  const int ty = threadIdx.x >> 5;   // 0..7
#pragma unroll
  for (int r = ty; r < 32; r += 8)
    tile[r][tx] = W[(size_t)(k0 + r) * N + n0 + tx];
  __syncthreads();
#pragma unroll
  for (int r = ty; r < 32; r += 8)
    Wt[(size_t)(n0 + r) * K + k0 + tx] = f32_to_bf16_rne(tile[tx][r]);
}

// ---------------------------------------------------------------------------
// Row softmax over bf16 scores, in place; 256 threads (8 wave32) per row.
__global__ __launch_bounds__(256) void softmax_rows_bf16(
    unsigned short* __restrict__ S, int cols)
{
  unsigned short* p = S + (size_t)blockIdx.x * cols;
  const int t = threadIdx.x;
  const int lane = t & 31, wv = t >> 5;

  float vals[8];
  float mx = -3.4e38f;
#pragma unroll
  for (int i = 0; i < 8; ++i) {
    const float v = bf16_to_f32(p[t + i * 256]);
    vals[i] = v;
    mx = fmaxf(mx, v);
  }
#pragma unroll
  for (int m = 16; m >= 1; m >>= 1) mx = fmaxf(mx, __shfl_xor(mx, m, 32));

  __shared__ float redm[8];
  __shared__ float reds[8];
  if (lane == 0) redm[wv] = mx;
  __syncthreads();
  float bm = redm[0];
#pragma unroll
  for (int w = 1; w < 8; ++w) bm = fmaxf(bm, redm[w]);

  float s = 0.f;
#pragma unroll
  for (int i = 0; i < 8; ++i) {
    vals[i] = __expf(vals[i] - bm);
    s += vals[i];
  }
#pragma unroll
  for (int m = 16; m >= 1; m >>= 1) s += __shfl_xor(s, m, 32);
  if (lane == 0) reds[wv] = s;
  __syncthreads();
  float bs = 0.f;
#pragma unroll
  for (int w = 0; w < 8; ++w) bs += reds[w];
  const float inv = 1.f / bs;

#pragma unroll
  for (int i = 0; i < 8; ++i) p[t + i * 256] = f32_to_bf16_rne(vals[i] * inv);
}

// ---------------------------------------------------------------------------
extern "C" void kernel_launch(void* const* d_in, const int* in_sizes, int n_in,
                              void* d_out, int out_size, void* d_ws, size_t ws_size,
                              hipStream_t stream) {
  (void)in_sizes; (void)n_in; (void)out_size; (void)ws_size;
  const float* x  = (const float*)d_in[0];   // [4,2048,1024]
  const float* Wq = (const float*)d_in[1];   // [1024,1024]
  const float* Wk = (const float*)d_in[2];
  const float* Wv = (const float*)d_in[3];
  float* out = (float*)d_out;                // [4,2048,1024]

  constexpr int  B  = 4, S = 2048, D = 1024;
  constexpr long MS = (long)B * S;           // 8192 folded rows

  // workspace layout (bf16 elements)
  unsigned short* xh  = (unsigned short*)d_ws;     // 16 MB
  unsigned short* Wtq = xh  + MS * D;              //  2 MB each
  unsigned short* Wtk = Wtq + (long)D * D;
  unsigned short* Wtv = Wtk + (long)D * D;
  unsigned short* Qb  = Wtv + (long)D * D;         // 16 MB
  unsigned short* Kb  = Qb  + MS * D;              // 16 MB
  unsigned short* Vt  = Kb  + MS * D;              // 16 MB  (transposed [b][d][key])
  unsigned short* Sb  = Vt  + MS * D;              // 32 MB  scores/probs

  const dim3 blk(256);

  // 0) fp32 -> bf16 conversions
  conv_f32_bf16<<<dim3((unsigned)(MS * D / 1024)), blk, 0, stream>>>(x, xh);
  {
    const dim3 g(D / 32, D / 32);
    transpose_conv_f32_bf16<<<g, blk, 0, stream>>>(Wq, Wtq, D, D);
    transpose_conv_f32_bf16<<<g, blk, 0, stream>>>(Wk, Wtk, D, D);
    transpose_conv_f32_bf16<<<g, blk, 0, stream>>>(Wv, Wtv, D, D);
  }

  // 1) Q,K projections (folded M=8192); V projection per-batch, transposed out
  {
    const dim3 g(D / BN, MS / BM, 1);
    gemm_bf16_wmma<1><<<g, blk, 0, stream>>>(xh, Wtq, Qb, D, D, D, D, 0, 0, 0, 1.0f);
    gemm_bf16_wmma<1><<<g, blk, 0, stream>>>(xh, Wtk, Kb, D, D, D, D, 0, 0, 0, 1.0f);
    const dim3 gv(D / BN, S / BM, B);
    gemm_bf16_wmma<2><<<gv, blk, 0, stream>>>(xh, Wtv, Vt, D, D, D, /*ldc=*/S,
                                              (long)S * D, 0, (long)D * S, 1.0f);
  }

  // 2) scores: S = Q @ K^T * 1/32  (B = K[key][d] is already [N][K])
  {
    const dim3 g(S / BN, S / BM, B);
    gemm_bf16_wmma<1><<<g, blk, 0, stream>>>(Qb, Kb, Sb, D, D, D, S,
                                             (long)S * D, (long)S * D,
                                             (long)S * S, 0.03125f);
  }

  // 3) softmax rows (8192 rows of 2048), in-place bf16
  softmax_rows_bf16<<<B * S, blk, 0, stream>>>(Sb, S);

  // 4) context: out = P @ V  (B = Vt[d][key] is [N][K]), f32 out
  {
    const dim3 g(D / BN, S / BM, B);
    gemm_bf16_wmma<0><<<g, blk, 0, stream>>>(Sb, Vt, out, S, S, S, D,
                                             (long)S * S, (long)D * S,
                                             (long)S * D, 1.0f);
  }
}
